// NNUE_52776558133968
// MI455X (gfx1250) — compile-verified, matrix operand order checked by reading, not run on hardware
//
#include <hip/hip_runtime.h>
#include <hip/hip_bf16.h>
#include <math.h>

typedef __attribute__((ext_vector_type(16))) _Float16 v16h;
typedef __attribute__((ext_vector_type(8)))  _Float16 v8h;
typedef __attribute__((ext_vector_type(4)))  _Float16 v4h;
typedef __attribute__((ext_vector_type(4)))  float    v4f;
typedef __attribute__((ext_vector_type(8)))  float    v8f;

#define HDIM 512      // hidden per perspective
#define KDIM 1024     // 2*HDIM (layer-1 K)
#define N1   128      // layer-1 out
#define N2   32       // layer-2 out
#define XS   1032     // LDS row stride (halves) for X tile, padded (+8) -> conflict-free
#define H1S  136      // LDS row stride (halves) for H1 tile, padded (+8)

__device__ __forceinline__ float clamp01(float v) {
  return fminf(fmaxf(v, 0.0f), 1.0f);
}

// ---------------------------------------------------------------------------
// Kernel 0: convert W1 (128x1024 f32) / W2 (32x128 f32) into f16 WMMA
// B-fragments, pre-swizzled so the GEMM loop does one contiguous 32B load per
// lane.  Fragment layout (per 16x32 B tile covering out-cols [t*16, t*16+16),
// K-slice [kc*32, kc*32+32)):
//   lane = nn + 16*hiF where hiF = (kk>>3)&1
//   half h = (kk&7) + (kk>=16 ? 8 : 0)
// ---------------------------------------------------------------------------
__global__ void nnue_pack_weights(const float* __restrict__ W1,
                                  const float* __restrict__ W2,
                                  _Float16* __restrict__ W1h,
                                  _Float16* __restrict__ W2h) {
  int tid = blockIdx.x * blockDim.x + threadIdx.x;
  if (tid < N1 * KDIM) {
    int n = tid / KDIM, k = tid - n * KDIM;
    int t = n >> 4, nn = n & 15;
    int kc = k >> 5, kk = k & 31;
    int hiF = (kk >> 3) & 1;
    int h   = (kk & 7) + ((kk & 16) ? 8 : 0);
    W1h[(((t * 32 + kc) * 32) + (nn + 16 * hiF)) * 16 + h] = (_Float16)W1[tid];
  } else if (tid < N1 * KDIM + N2 * N1) {
    int e = tid - N1 * KDIM;
    int n = e / N1, k = e - n * N1;     // W2: [32][128]
    int t = n >> 4, nn = n & 15;
    int kc = k >> 5, kk = k & 31;
    int hiF = (kk >> 3) & 1;
    int h   = (kk & 7) + ((kk & 16) ? 8 : 0);
    W2h[(((t * 4 + kc) * 32) + (nn + 16 * hiF)) * 16 + h] = (_Float16)W2[e];
  }
}

// ---------------------------------------------------------------------------
// Fused NNUE kernel: one workgroup (8 wave32) handles 16 positions.
// ---------------------------------------------------------------------------
__global__ __launch_bounds__(256)
void nnue_fused(const int* __restrict__ stm_idx, const int* __restrict__ stm_off,
                const int* __restrict__ nstm_idx, const int* __restrict__ nstm_off,
                const float* __restrict__ emb,
                const float* __restrict__ b1, const float* __restrict__ b2,
                const float* __restrict__ W3, const float* __restrict__ b3,
                const _Float16* __restrict__ W1h, const _Float16* __restrict__ W2h,
                float* __restrict__ out, int Btot, int totalIdx) {
  __shared__ _Float16 Xlds[16 * XS];     // [pos][K=0..1023] f16, padded rows
  __shared__ _Float16 H1lds[16 * H1S];   // [pos][128] f16, padded rows
  __shared__ float    H2lds[16 * N2];    // [pos][32] f32

  const int tid    = threadIdx.x;
  const int wave   = tid >> 5;
  const int lane   = tid & 31;
  const int wgPos0 = blockIdx.x * 16;

  // ---------------- Phase A: embedding-bag gather+sum ----------------------
  // Each wave owns 2 positions; both perspectives of a position are gathered
  // concurrently (2 independent index streams -> 8 independent b128 loads per
  // iteration; unroll 2 -> up to 16 outstanding loads to hide L2 latency).
  // Lane l accumulates float4 slices at row-offset (j*32+l)*4, j=0..3: each
  // load instruction covers a contiguous 512B slice of an emb row.
  for (int bb = 0; bb < 2; ++bb) {
    const int pos = wave * 2 + bb;       // 0..15
    const int p   = wgPos0 + pos;

    const int start0 = stm_off[p];
    const int end0   = (p + 1 < Btot) ? stm_off[p + 1] : totalIdx;
    const int cnt0   = end0 - start0;
    const int start1 = nstm_off[p];
    const int end1   = (p + 1 < Btot) ? nstm_off[p + 1] : totalIdx;
    const int cnt1   = end1 - start1;

    int idx0 = 0, idx1 = 0;
    if (lane < cnt0) idx0 = stm_idx[start0 + lane];
    if (lane < cnt1) idx1 = nstm_idx[start1 + lane];

    v4f sa0 = {}, sa1 = {}, sa2 = {}, sa3 = {};   // stm accumulators
    v4f sb0 = {}, sb1 = {}, sb2 = {}, sb3 = {};   // nstm accumulators

    const int cmax = cnt0 > cnt1 ? cnt0 : cnt1;
    #pragma unroll 2
    for (int i = 0; i < cmax; ++i) {
      if (i < cnt0) {                          // wave-uniform branch
        const int row = __builtin_amdgcn_readlane(idx0, i);
        const v4f* r4 = (const v4f*)(emb + (size_t)row * HDIM);
        sa0 += r4[lane];
        sa1 += r4[32 + lane];
        sa2 += r4[64 + lane];
        sa3 += r4[96 + lane];
      }
      if (i < cnt1) {                          // wave-uniform branch
        const int row = __builtin_amdgcn_readlane(idx1, i);
        const v4f* r4 = (const v4f*)(emb + (size_t)row * HDIM);
        sb0 += r4[lane];
        sb1 += r4[32 + lane];
        sb2 += r4[64 + lane];
        sb3 += r4[96 + lane];
      }
    }

    _Float16* xr0 = &Xlds[pos * XS];           // stm half  (K 0..511)
    _Float16* xr1 = xr0 + HDIM;                // nstm half (K 512..1023)
    v4f acc[8] = {sa0, sa1, sa2, sa3, sb0, sb1, sb2, sb3};
    #pragma unroll
    for (int j = 0; j < 4; ++j) {
      v4h h0, h1v;
      h0.x  = (_Float16)clamp01(acc[j].x);
      h0.y  = (_Float16)clamp01(acc[j].y);
      h0.z  = (_Float16)clamp01(acc[j].z);
      h0.w  = (_Float16)clamp01(acc[j].w);
      h1v.x = (_Float16)clamp01(acc[4 + j].x);
      h1v.y = (_Float16)clamp01(acc[4 + j].y);
      h1v.z = (_Float16)clamp01(acc[4 + j].z);
      h1v.w = (_Float16)clamp01(acc[4 + j].w);
      *(v4h*)(&xr0[(j * 32 + lane) * 4]) = h0;
      *(v4h*)(&xr1[(j * 32 + lane) * 4]) = h1v;
    }
  }
  __syncthreads();

  // ---------------- Phase B: layer 1 GEMM [16,1024]x[1024,128] --------------
  // Wave w owns out-cols [16w,16w+16).  A-fragment per ISA 16-bit 16x32 table:
  // lane-lo holds K {0..7,16..23}, lane-hi holds K {8..15,24..31}.
  const int m   = lane & 15;
  const int hiF = lane >> 4;
  {
    v8f c1 = {};
    const _Float16* xrow = &Xlds[m * XS];
    for (int kc = 0; kc < KDIM / 32; ++kc) {
      const int k0 = kc * 32;
      v8h alo = *(const v8h*)(&xrow[k0 + hiF * 8]);
      v8h ahi = *(const v8h*)(&xrow[k0 + 16 + hiF * 8]);
      v16h a = __builtin_shufflevector(alo, ahi, 0, 1, 2, 3, 4, 5, 6, 7,
                                       8, 9, 10, 11, 12, 13, 14, 15);
      v16h b = *(const v16h*)(&W1h[(((wave * 32 + kc) * 32) + lane) * 16]);
      c1 = __builtin_amdgcn_wmma_f32_16x16x32_f16(
          false, a, false, b, (short)0, c1, false, false);
    }
    const float bv = b1[wave * 16 + m];
    #pragma unroll
    for (int r = 0; r < 8; ++r) {
      const int M = r + 8 * hiF;      // C/D layout: VGPR r, lanes 16-31 -> M+8
      H1lds[M * H1S + wave * 16 + m] = (_Float16)clamp01(c1[r] + bv);
    }
  }
  __syncthreads();

  // ---------------- Phase C: layer 2 GEMM [16,128]x[128,32] -----------------
  if (wave < 2) {
    v8f c2 = {};
    const _Float16* hrow = &H1lds[m * H1S];
    #pragma unroll
    for (int kc = 0; kc < 4; ++kc) {
      const int k0 = kc * 32;
      v8h alo = *(const v8h*)(&hrow[k0 + hiF * 8]);
      v8h ahi = *(const v8h*)(&hrow[k0 + 16 + hiF * 8]);
      v16h a = __builtin_shufflevector(alo, ahi, 0, 1, 2, 3, 4, 5, 6, 7,
                                       8, 9, 10, 11, 12, 13, 14, 15);
      v16h b = *(const v16h*)(&W2h[(((wave * 4 + kc) * 32) + lane) * 16]);
      c2 = __builtin_amdgcn_wmma_f32_16x16x32_f16(
          false, a, false, b, (short)0, c2, false, false);
    }
    const float bv = b2[wave * 16 + m];
    #pragma unroll
    for (int r = 0; r < 8; ++r) {
      const int M = r + 8 * hiF;
      H2lds[M * N2 + wave * 16 + m] = clamp01(c2[r] + bv);
    }
  }
  __syncthreads();

  // ---------------- Phase D: layer 3 [16,32]x[32,1] + tanh ------------------
  if (tid < 16) {
    float s = b3[0];
    #pragma unroll
    for (int j = 0; j < N2; ++j) s += H2lds[tid * N2 + j] * W3[j];
    out[wgPos0 + tid] = tanhf(s);
  }
}

extern "C" void kernel_launch(void* const* d_in, const int* in_sizes, int n_in,
                              void* d_out, int out_size, void* d_ws, size_t ws_size,
                              hipStream_t stream) {
  const int*   stm_idx  = (const int*)d_in[0];
  const int*   stm_off  = (const int*)d_in[1];
  const int*   nstm_idx = (const int*)d_in[2];
  const int*   nstm_off = (const int*)d_in[3];
  const float* emb      = (const float*)d_in[4];
  const float* W1       = (const float*)d_in[5];
  const float* b1       = (const float*)d_in[6];
  const float* W2       = (const float*)d_in[7];
  const float* b2       = (const float*)d_in[8];
  const float* W3       = (const float*)d_in[9];
  const float* b3       = (const float*)d_in[10];
  float*       out      = (float*)d_out;

  const int totalIdx = in_sizes[0];   // B*L
  const int Btot     = in_sizes[1];   // B (offsets array length)

  _Float16* W1h = (_Float16*)d_ws;                                  // 256 KB
  _Float16* W2h = (_Float16*)((char*)d_ws + (size_t)N1 * KDIM * 2); // 8 KB

  const int packElems = N1 * KDIM + N2 * N1;
  nnue_pack_weights<<<(packElems + 255) / 256, 256, 0, stream>>>(W1, W2, W1h, W2h);

  nnue_fused<<<Btot / 16, 256, 0, stream>>>(
      stm_idx, stm_off, nstm_idx, nstm_off, emb, b1, b2, W3, b3,
      W1h, W2h, out, Btot, totalIdx);
}
